// VolumnRenderer_81956565942378
// MI455X (gfx1250) — matrix operand order repaired and checked by reading, not compiled, not run on hardware
//
#include <hip/hip_runtime.h>

typedef __attribute__((ext_vector_type(16))) _Float16 v16h;
typedef __attribute__((ext_vector_type(8)))  float    v8f;

#define N_RAYS    65536
#define N_SAMPLES 2097152
#define HID       64
#define N_TILES   (N_SAMPLES / 16)

// ---- weight fragment layout (halfs); fragment = 512 halfs --------------
//   lane l holds 16 halfs; elem j -> K = kt*32 + (l<16 ? j : 16+j),
//                                   N = nt*16 + (l&15)
#define W1_OFF    0        // 2kt x 4nt = 8 frags (K=63 padded)
#define W2_OFF    4096     // 8 frags
#define WF_OFF    8192     // 8 frags
#define WR1_OFF   12288    // 3kt x 4nt = 12 frags (K=91 padded)
#define WDH_OFF   18432    // sigma head [64x16], col0=Wd : 2 frags
#define WR2H_OFF  19456    // rgb head   [64x16], col0..2=Wr2 : 2 frags
#define FRAG_HALFS 20480

// ---- f32 param block (floats) ------------------------------------------
#define F_B1   0
#define F_B2   64
#define F_BF   128
#define F_BR1  192
#define F_BD   256
#define F_BR2  257
#define FLT_TOTAL 260

// ---- workspace layout (bytes) ------------------------------------------
#define WS_FRAG_B  0
#define WS_FLT_B   40960
#define WS_RGBA_B  65536   // float4/sample = 33,554,432 B

// ========================================================================
// Kernel 0: convert + pre-swizzle weights into WMMA B-fragment layout
// ========================================================================
__global__ void nerf_prep_kernel(const float* __restrict__ W1,
                                 const float* __restrict__ W2,
                                 const float* __restrict__ Wf,
                                 const float* __restrict__ Wr1,
                                 const float* __restrict__ b1,
                                 const float* __restrict__ b2,
                                 const float* __restrict__ bf,
                                 const float* __restrict__ br1,
                                 const float* __restrict__ Wd,
                                 const float* __restrict__ bd,
                                 const float* __restrict__ Wr2,
                                 const float* __restrict__ br2,
                                 _Float16* __restrict__ wsFrag,
                                 float* __restrict__ wsFlt) {
  int idx = blockIdx.x * blockDim.x + threadIdx.x;
  if (idx < FRAG_HALFS) {
    const float* M = nullptr;
    int Krows = 64, r = 0, region = 0;   // region 0=dense, 1=Wd head, 2=Wr2 head
    if (idx < W2_OFF)        { M = W1;  Krows = 63; r = idx - W1_OFF;  }
    else if (idx < WF_OFF)   { M = W2;              r = idx - W2_OFF;  }
    else if (idx < WR1_OFF)  { M = Wf;              r = idx - WF_OFF;  }
    else if (idx < WDH_OFF)  { M = Wr1; Krows = 91; r = idx - WR1_OFF; }
    else if (idx < WR2H_OFF) { region = 1;          r = idx - WDH_OFF; }
    else                     { region = 2;          r = idx - WR2H_OFF;}
    int fi = r >> 9, e = r & 511, lane = e >> 4, j = e & 15;
    int kt, nt;
    if (region == 0) { kt = fi >> 2; nt = fi & 3; } else { kt = fi; nt = 0; }
    int K = kt * 32 + ((lane < 16) ? j : 16 + j);
    int c = lane & 15;
    int N = nt * 16 + c;
    float v;
    if (region == 0)      v = (K < Krows) ? M[K * HID + N] : 0.0f;
    else if (region == 1) v = (c == 0 && K < 64) ? Wd[K] : 0.0f;
    else                  v = (c < 3 && K < 64) ? Wr2[K * 3 + c] : 0.0f;
    wsFrag[idx] = (_Float16)v;
  } else if (idx < FRAG_HALFS + FLT_TOTAL) {
    int j = idx - FRAG_HALFS;
    float v;
    if      (j < 64)   v = b1[j];
    else if (j < 128)  v = b2[j - 64];
    else if (j < 192)  v = bf[j - 128];
    else if (j < 256)  v = br1[j - 192];
    else if (j == 256) v = bd[0];
    else               v = br2[j - 257];
    wsFlt[j] = v;
  }
}

// ========================================================================
// fragment loaders (contiguous k order, same in A and B)
// ========================================================================
__device__ __forceinline__ v16h load_b_frag(const _Float16* fbase, int lane) {
  union { v16h v; uint4 q[2]; } u;
  const uint4* p = (const uint4*)(fbase + lane * 16);
  u.q[0] = p[0];
  u.q[1] = p[1];
  return u.v;
}

// A (16x32 f16): lane<16 holds K 0..15 ; lane>=16 holds K 16..31
__device__ __forceinline__ v16h load_a_frag(const _Float16* act, int stride,
                                            int kt, int lane) {
  int m  = lane & 15;
  int k0 = kt * 32 + ((lane >> 4) << 4);
  const _Float16* r = act + m * stride + k0;
  union { v16h v; uint4 q[2]; } u;
  u.q[0] = *(const uint4*)(r);
  u.q[1] = *(const uint4*)(r + 8);
  return u.v;
}

// dense layer: [16 x 32*KT] @ [32*KT x 64] (+bias, optional relu)
__device__ __forceinline__ void layer_wmma(const _Float16* actIn, int strideIn,
                                           int KT, const _Float16* wbase,
                                           const float* bias,
                                           _Float16* actOut, int strideOut,
                                           int lane, bool do_relu) {
#pragma unroll
  for (int nt = 0; nt < 4; ++nt) {
    v8f c = {};
    float b = bias[nt * 16 + (lane & 15)];
    for (int kt = 0; kt < KT; ++kt) {
      v16h a = load_a_frag(actIn, strideIn, kt, lane);
      v16h w = load_b_frag(wbase + (kt * 4 + nt) * 512, lane);
      c = __builtin_amdgcn_wmma_f32_16x16x32_f16(false, a, false, w,
                                                 (short)0, c, false, false);
    }
    int rbase = (lane < 16) ? 0 : 8;
    int col   = nt * 16 + (lane & 15);
#pragma unroll
    for (int r = 0; r < 8; ++r) {
      float v = c[r] + b;
      if (do_relu) v = fmaxf(v, 0.0f);
      actOut[(rbase + r) * strideOut + col] = (_Float16)v;
    }
  }
}

// head GEMM: [16 x 64] @ [64 x 16]
__device__ __forceinline__ v8f head_gemm(const _Float16* actIn, int strideIn,
                                         const _Float16* wbase, int lane) {
  v8f c = {};
#pragma unroll
  for (int kt = 0; kt < 2; ++kt) {
    v16h a = load_a_frag(actIn, strideIn, kt, lane);
    v16h w = load_b_frag(wbase + kt * 512, lane);
    c = __builtin_amdgcn_wmma_f32_16x16x32_f16(false, a, false, w,
                                               (short)0, c, false, false);
  }
  return c;
}

// ========================================================================
// Kernel 1: MLP over all samples (WMMA), writes float4 {r,g,b,alpha}
// ========================================================================
__global__ void __launch_bounds__(128)
nerf_mlp_kernel(const float* __restrict__ rays_o,
                const float* __restrict__ viewdirs,
                const float* __restrict__ t_starts,
                const float* __restrict__ t_ends,
                const int* __restrict__ ray_indices,
                const _Float16* __restrict__ wsFrag,
                const float* __restrict__ wsFlt,
                float4* __restrict__ rgba) {
  __shared__ __align__(16) _Float16 sW[FRAG_HALFS];          // 40960 B
  __shared__ float sF[FLT_TOTAL];                            //  1040 B
  __shared__ __align__(16) _Float16 sAct0[4][16 * 64];       //  8192 B
  __shared__ __align__(16) _Float16 sAct1[4][16 * 96];       // 12288 B
  __shared__ float sHead[4][16][4];                          //  1024 B

  int tid = threadIdx.x;
  // stage weights via async global->LDS DMA (ASYNCcnt)
  {
    uint4* l = (uint4*)sW;
    const uint4* g = (const uint4*)wsFrag;
    for (int i = tid; i < FRAG_HALFS / 8; i += 128) {
      unsigned ldsa = (unsigned)(unsigned long long)(l + i);
      asm volatile("global_load_async_to_lds_b128 %0, %1, off"
                   :: "v"(ldsa), "v"(g + i) : "memory");
    }
    for (int i = tid; i < FLT_TOTAL; i += 128) sF[i] = wsFlt[i];
    asm volatile("s_wait_asynccnt 0x0" ::: "memory");
  }
  __syncthreads();

  int wave = tid >> 5;
  int lane = tid & 31;
  int sl   = lane & 15;     // sample slot in tile
  int hv   = lane >> 4;     // which half of the feature range this lane owns
  _Float16* buf0 = &sAct0[wave][0];   // stride 64
  _Float16* buf1 = &sAct1[wave][0];   // stride 96

  int tile_stride = gridDim.x * 4;
  for (int tile = blockIdx.x * 4 + wave; tile < N_TILES; tile += tile_stride) {
    int s = tile * 16 + sl;
    int snext = s + tile_stride * 16;
    if (snext < N_SAMPLES) __builtin_prefetch(&ray_indices[snext], 0, 3);

    int ri = ray_indices[s];
    float ox = rays_o[ri * 3 + 0], oy = rays_o[ri * 3 + 1], oz = rays_o[ri * 3 + 2];
    float dx = viewdirs[ri * 3 + 0], dy = viewdirs[ri * 3 + 1], dz = viewdirs[ri * 3 + 2];
    float ts = t_starts[s], te = t_ends[s];
    float tm = 0.5f * (ts + te);
    float p[3]  = {ox + dx * tm, oy + dy * tm, oz + dz * tm};
    float dv[3] = {dx, dy, dz};

    // positional encoding (L=10) -> buf0 cols [0,64), branch-free sincos
#pragma unroll
    for (int t = 0; t < 15; ++t) {
      int L = hv * 5 + t / 3;         // runtime hv, compile-time t/3
      const int c = t % 3;            // compile-time
      float ang = p[c] * (float)(1 << L);
      float sn, cn;
      __sincosf(ang, &sn, &cn);
      int base = sl * 64 + 3 + L * 6 + c;
      buf0[base]     = (_Float16)sn;
      buf0[base + 3] = (_Float16)cn;
    }
    if (hv == 0) {
      buf0[sl * 64 + 0] = (_Float16)p[0];
      buf0[sl * 64 + 1] = (_Float16)p[1];
      buf0[sl * 64 + 2] = (_Float16)p[2];
    } else {
      buf0[sl * 64 + 63] = (_Float16)0.0f;
    }
    // dir encoding (L=4) -> buf1 cols [64,96)
#pragma unroll
    for (int t = 0; t < 6; ++t) {
      int L = hv * 2 + t / 3;
      const int c = t % 3;
      float ang = dv[c] * (float)(1 << L);
      float sn, cn;
      __sincosf(ang, &sn, &cn);
      int base = sl * 96 + 64 + 3 + L * 6 + c;
      buf1[base]     = (_Float16)sn;
      buf1[base + 3] = (_Float16)cn;
    }
    if (hv == 0) {
      buf1[sl * 96 + 64] = (_Float16)dv[0];
      buf1[sl * 96 + 65] = (_Float16)dv[1];
      buf1[sl * 96 + 66] = (_Float16)dv[2];
    } else {
#pragma unroll
      for (int c2 = 91; c2 < 96; ++c2) buf1[sl * 96 + c2] = (_Float16)0.0f;
    }

    // h1 = relu(enc @ W1 + b1)            buf0 -> buf1[0:64]
    layer_wmma(buf0, 64, 2, sW + W1_OFF, sF + F_B1, buf1, 96, lane, true);
    // h2 = relu(h1 @ W2 + b2)             buf1 -> buf0
    layer_wmma(buf1, 96, 2, sW + W2_OFF, sF + F_B2, buf0, 64, lane, true);

    // sigma head via WMMA: h2 @ [Wd|0..]  (result col 0)
    {
      v8f cs8 = head_gemm(buf0, 64, sW + WDH_OFF, lane);
      if ((lane & 15) == 0) {
        int rbase = (lane < 16) ? 0 : 8;
        float bd0 = sF[F_BD];
#pragma unroll
        for (int r = 0; r < 8; ++r) sHead[wave][rbase + r][3] = cs8[r] + bd0;
      }
    }

    // feat = h2 @ Wf + bf (no relu)       buf0 -> buf1[0:64] (direnc intact)
    layer_wmma(buf0, 64, 2, sW + WF_OFF, sF + F_BF, buf1, 96, lane, false);
    // hd = relu([feat,direnc] @ Wr1+br1)  buf1(96) -> buf0
    layer_wmma(buf1, 96, 3, sW + WR1_OFF, sF + F_BR1, buf0, 64, lane, true);

    // rgb head via WMMA: hd @ [Wr2|0..]   (result cols 0..2)
    {
      v8f cr8 = head_gemm(buf0, 64, sW + WR2H_OFF, lane);
      int cc = lane & 15;
      if (cc < 3) {
        int rbase = (lane < 16) ? 0 : 8;
        float bb = sF[F_BR2 + cc];
#pragma unroll
        for (int r = 0; r < 8; ++r) sHead[wave][rbase + r][cc] = cr8[r] + bb;
      }
    }

    // finalize + output (lanes 0..15)
    if (lane < 16) {
      float sg    = fmaxf(sHead[wave][sl][3], 0.0f);
      float alpha = 1.0f - __expf(-sg * (te - ts));
      float4 o4;
      o4.x = 1.0f / (1.0f + __expf(-sHead[wave][sl][0]));
      o4.y = 1.0f / (1.0f + __expf(-sHead[wave][sl][1]));
      o4.z = 1.0f / (1.0f + __expf(-sHead[wave][sl][2]));
      o4.w = alpha;
      rgba[s] = o4;
    }
  }
}

// ========================================================================
// Kernel 2: per-ray compositing (ray_indices sorted -> contiguous segments)
// ========================================================================
__device__ __forceinline__ int lower_bound_i32(const int* a, int n, int key) {
  int lo = 0, hi = n;
  while (lo < hi) {
    int mid = (lo + hi) >> 1;
    if (a[mid] < key) lo = mid + 1; else hi = mid;
  }
  return lo;
}

__global__ void nerf_composite_kernel(const float* __restrict__ t_starts,
                                      const float* __restrict__ t_ends,
                                      const int* __restrict__ ray_indices,
                                      const float4* __restrict__ rgba,
                                      float* __restrict__ out) {
  int r = blockIdx.x * blockDim.x + threadIdx.x;
  if (r >= N_RAYS) return;
  int beg = lower_bound_i32(ray_indices, N_SAMPLES, r);
  int end = lower_bound_i32(ray_indices, N_SAMPLES, r + 1);

  float T = 1.0f;
  float cr = 0.0f, cg = 0.0f, cb = 0.0f, opac = 0.0f, depth = 0.0f;
  for (int j = beg; j < end; ++j) {
    float4 v = rgba[j];
    float a = v.w;
    float w = T * a;
    float tm = 0.5f * (t_starts[j] + t_ends[j]);
    cr += w * v.x; cg += w * v.y; cb += w * v.z;
    opac += w;
    depth += w * tm;
    T *= 1.0f - fminf(a, 1.0f - 1e-7f);
  }
  float bk = 1.0f - opac;   // white background
  out[r * 3 + 0] = cr + bk;
  out[r * 3 + 1] = cg + bk;
  out[r * 3 + 2] = cb + bk;
  out[N_RAYS * 3 + r] = opac;
  out[N_RAYS * 4 + r] = depth;
}

// ========================================================================
extern "C" void kernel_launch(void* const* d_in, const int* in_sizes, int n_in,
                              void* d_out, int out_size, void* d_ws, size_t ws_size,
                              hipStream_t stream) {
  const float* rays_o      = (const float*)d_in[0];
  const float* viewdirs    = (const float*)d_in[1];
  const float* t_starts    = (const float*)d_in[2];
  const float* t_ends      = (const float*)d_in[3];
  const float* W1          = (const float*)d_in[4];
  const float* b1          = (const float*)d_in[5];
  const float* W2          = (const float*)d_in[6];
  const float* b2          = (const float*)d_in[7];
  const float* Wd          = (const float*)d_in[8];
  const float* bd          = (const float*)d_in[9];
  const float* Wf          = (const float*)d_in[10];
  const float* bf          = (const float*)d_in[11];
  const float* Wr1         = (const float*)d_in[12];
  const float* br1         = (const float*)d_in[13];
  const float* Wr2         = (const float*)d_in[14];
  const float* br2         = (const float*)d_in[15];
  const int*   ray_indices = (const int*)d_in[16];

  _Float16* wsFrag = (_Float16*)((char*)d_ws + WS_FRAG_B);
  float*    wsFlt  = (float*)((char*)d_ws + WS_FLT_B);
  float4*   rgba   = (float4*)((char*)d_ws + WS_RGBA_B);
  float*    out    = (float*)d_out;

  int prep_threads = FRAG_HALFS + FLT_TOTAL;
  nerf_prep_kernel<<<(prep_threads + 127) / 128, 128, 0, stream>>>(
      W1, W2, Wf, Wr1, b1, b2, bf, br1, Wd, bd, Wr2, br2, wsFrag, wsFlt);

  nerf_mlp_kernel<<<2048, 128, 0, stream>>>(
      rays_o, viewdirs, t_starts, t_ends, ray_indices, wsFrag, wsFlt, rgba);

  nerf_composite_kernel<<<N_RAYS / 256, 256, 0, stream>>>(
      t_starts, t_ends, ray_indices, rgba, out);
}
